// MultiHeadAttentionFLAX_89455578841393
// MI455X (gfx1250) — matrix operand order repaired and checked by reading, not compiled, not run on hardware
//
#include <hip/hip_runtime.h>

// Problem constants (fixed by the reference): B=2, S=2048, D=1024, H=16, HD=64
#define BB 2
#define SS 2048
#define DD 1024
#define HH 16
#define HDm 64
#define MM (BB*SS)          // 4096 rows
#define SCALE 0.125f        // 64^-0.5

typedef __attribute__((ext_vector_type(16))) __bf16 v16bf;
typedef __attribute__((ext_vector_type(8)))  float  v8f;

union FragAB { v16bf v; unsigned int u[8]; };
union FragC  { v8f   v; float        f[8]; };

__device__ __forceinline__ unsigned short f32_to_bf16_bits(float x) {
  unsigned int u = __float_as_uint(x);
  u += 0x7FFFu + ((u >> 16) & 1u);   // round-to-nearest-even
  return (unsigned short)(u >> 16);
}

// CDNA5 async global->LDS copy (ASYNCcnt-tracked). VDST = LDS byte address,
// VADDR = 64-bit global address. Flat shared addresses truncate to the LDS
// offset (addr[31:0]) per the gfx1250 aperture rules.
__device__ __forceinline__ void async_copy_b128(unsigned lds_off, const void* gptr) {
  asm volatile("global_load_async_to_lds_b128 %0, %1, off"
               :: "v"(lds_off), "v"(gptr) : "memory");
}
__device__ __forceinline__ unsigned lds_addr_of(const void* p) {
  return (unsigned)(unsigned long long)(size_t)p;
}
__device__ __forceinline__ void wait_async0() {
  asm volatile("s_wait_asynccnt 0x0" ::: "memory");
}

// ---------------------------------------------------------------- converts
__global__ void cvt_f32_to_bf16(const float* __restrict__ in,
                                unsigned short* __restrict__ out, int n) {
  int i = blockIdx.x * blockDim.x + threadIdx.x;
  if (i < n) out[i] = f32_to_bf16_bits(in[i]);
}

// out[n*DD + k] = bf16(in[k*DD + n])  (store W transposed: rows = N, cols = K)
__global__ void cvt_transpose_w(const float* __restrict__ in,
                                unsigned short* __restrict__ out) {
  int i = blockIdx.x * blockDim.x + threadIdx.x;
  if (i >= DD * DD) return;
  int nn = i >> 10;          // / DD
  int k  = i & (DD - 1);
  out[i] = f32_to_bf16_bits(in[k * DD + nn]);
}

// ---------------------------------------------------------------- GEMM
// C[MM x DD] = A[MM x DD](bf16, row-major) * Wt[DD x DD](bf16, B^T row-major) + bias
// B-tile (64 x 32 bf16, shared by all 4 waves) is staged into LDS with
// double-buffered async copies so the copy of k-block n+1 overlaps the WMMAs
// of k-block n.
// mode 0: bf16 out row-major          (Q, K)
// mode 1: bf16 out per-head transpose (V  -> Vt[b][h][hd][s])
// mode 2: f32  out row-major          (final projection)
#define BROW 80   // LDS row stride (64B data + 16B pad), 16B aligned
__global__ void gemm_wmma(const unsigned short* __restrict__ A,
                          const unsigned short* __restrict__ Wt,
                          const float*          __restrict__ bias,
                          unsigned short*       __restrict__ obf,
                          float*                __restrict__ of32,
                          int mode) {
  __shared__ __align__(16) unsigned char Bsm[2][64 * BROW];

  const int tid   = threadIdx.x;
  const int lane  = tid & 31;
  const int wave  = tid >> 5;
  const int khalf = lane >> 4;        // which 16-lane half
  const int l16   = lane & 15;
  const int m0 = blockIdx.x * 64 + wave * 16;
  const int n0 = blockIdx.y * 64;

  const unsigned char* Wb = (const unsigned char*)Wt;
  auto issueB = [&](int k0, int buf) {
    // 64 rows x 64B = 256 x 16B chunks, 2 per thread
#pragma unroll
    for (int j = 0; j < 2; ++j) {
      int c   = tid * 2 + j;        // 0..255
      int row = c >> 2;             // 0..63
      int off = (c & 3) * 16;       // 0..48
      unsigned lds = lds_addr_of(&Bsm[buf][row * BROW + off]);
      const void* g = Wb + ((size_t)(n0 + row) * DD + k0) * 2 + off;
      async_copy_b128(lds, g);
    }
  };

  FragC acc[4];
#pragma unroll
  for (int t = 0; t < 4; ++t)
#pragma unroll
    for (int i = 0; i < 8; ++i) acc[t].f[i] = 0.0f;

  const int arow = m0 + l16;
  const unsigned short* Arow = A + (size_t)arow * DD;

  issueB(0, 0);
  wait_async0();
  __syncthreads();

  for (int k0 = 0; k0 < DD; k0 += 32) {
    const int cur = (k0 >> 5) & 1;
    if (k0 + 32 < DD) issueB(k0 + 32, cur ^ 1);

    __builtin_prefetch(Arow + k0 + 64, 0, 3);   // global_prefetch_b8
    // A fragment: 16x32 bf16. Lanes 0-15: K 0-15 ; lanes 16-31: K 16-31
    FragAB a;
#pragma unroll
    for (int i = 0; i < 8; ++i) {
      int kk = k0 + 16 * khalf + 2 * i;
      a.u[i] = *(const unsigned int*)(Arow + kk);
    }
#pragma unroll
    for (int t = 0; t < 4; ++t) {
      // B fragment from LDS: lanes 0-15 hold K 0-15, lanes 16-31 hold K 16-31.
      FragAB bfr;
      const unsigned char* bp = &Bsm[cur][(t * 16 + l16) * BROW + 32 * khalf];
#pragma unroll
      for (int i = 0; i < 8; ++i)
        bfr.u[i] = *(const unsigned int*)(bp + 4 * i);
      acc[t].v = __builtin_amdgcn_wmma_f32_16x16x32_bf16(
          false, a.v, false, bfr.v, (short)0, acc[t].v, false, false);
    }

    wait_async0();
    __syncthreads();
  }

#pragma unroll
  for (int t = 0; t < 4; ++t) {
    int nn = n0 + t * 16 + l16;
    float bn = bias[nn];
#pragma unroll
    for (int i = 0; i < 8; ++i) {
      int   m   = m0 + i + 8 * khalf;   // C layout: VGPR i -> row i (lanes 0-15) / i+8
      float val = acc[t].f[i] + bn;
      if (mode == 2) {
        of32[(size_t)m * DD + nn] = val;
      } else if (mode == 1) {
        int b = m >> 11, s = m & (SS - 1);       // m = b*SS + s
        int h = nn >> 6, hd = nn & (HDm - 1);    // nn = h*HD + hd
        obf[(((size_t)(b * HH + h)) * HDm + hd) * SS + s] = f32_to_bf16_bits(val);
      } else {
        obf[(size_t)m * DD + nn] = f32_to_bf16_bits(val);
      }
    }
  }
}

// ---------------------------------------------------------------- attention
// One wave = 16 query rows of one (b,h); 4 waves/block share (b,h).
// K-tile (32 keys x 64) and V-tile (64 hd x 32 keys) are staged into LDS with
// double-buffered async copies shared by the whole block.
// Q,K: bf16 [b][s][h*HD+hd] (row-major, stride DD). Vt: bf16 [b][h][hd][s].
// O: bf16 [b][s][h*HD+hd].
#define KROW 144  // 128B data + 16B pad
#define VROW 80   // 64B data + 16B pad
__global__ void attn_wmma(const unsigned short* __restrict__ Q,
                          const unsigned short* __restrict__ Kb,
                          const unsigned short* __restrict__ Vt,
                          unsigned short*       __restrict__ O) {
  __shared__ __align__(16) unsigned char  Ksm[2][32 * KROW];
  __shared__ __align__(16) unsigned char  Vsm[2][64 * VROW];
  __shared__ __align__(16) unsigned short Plds[4][16 * 34];  // stride 17 dwords/row

  const int tid   = threadIdx.x;
  const int lane  = tid & 31;
  const int wave  = tid >> 5;
  const int khalf = lane >> 4;
  const int l16   = lane & 15;
  const int bh = blockIdx.y;
  const int b  = bh >> 4, h = bh & 15;
  const int q0 = blockIdx.x * 64 + wave * 16;

  const size_t headOff = ((size_t)b * SS) * DD + (size_t)h * HDm;
  const unsigned char* Kg = (const unsigned char*)Kb;
  const unsigned char* Vg = (const unsigned char*)Vt;

  auto issueKV = [&](int kbase, int buf) {
    // K tile: 32 rows x 128B = 256 x 16B chunks
#pragma unroll
    for (int j = 0; j < 2; ++j) {
      int c   = tid * 2 + j;
      int row = c >> 3;             // key 0..31
      int off = (c & 7) * 16;       // 0..112
      unsigned lds = lds_addr_of(&Ksm[buf][row * KROW + off]);
      const void* g = Kg + (headOff + (size_t)(kbase + row) * DD) * 2 + off;
      async_copy_b128(lds, g);
    }
    // V tile: 64 rows x 64B = 256 x 16B chunks
#pragma unroll
    for (int j = 0; j < 2; ++j) {
      int c   = tid * 2 + j;
      int row = c >> 2;             // hd 0..63
      int off = (c & 3) * 16;       // 0..48
      unsigned lds = lds_addr_of(&Vsm[buf][row * VROW + off]);
      const void* g = Vg + (((size_t)(b * HH + h) * HDm + row) * SS + kbase) * 2 + off;
      async_copy_b128(lds, g);
    }
  };

  // Q fragments: 16 rows x 64 (two K32 chunks), loaded once.
  FragAB qa[2];
  {
    int s = q0 + l16;
#pragma unroll
    for (int c = 0; c < 2; ++c)
#pragma unroll
      for (int i = 0; i < 8; ++i) {
        int kk = 32 * c + 16 * khalf + 2 * i;
        qa[c].u[i] = *(const unsigned int*)(Q + headOff + (size_t)s * DD + kk);
      }
  }

  float mrow[8], lrow[8];
  FragC acc[4];
#pragma unroll
  for (int i = 0; i < 8; ++i) { mrow[i] = -1e30f; lrow[i] = 0.0f; }
#pragma unroll
  for (int t = 0; t < 4; ++t)
#pragma unroll
    for (int i = 0; i < 8; ++i) acc[t].f[i] = 0.0f;

  issueKV(0, 0);
  wait_async0();
  __syncthreads();

  for (int kb = 0; kb < SS / 32; ++kb) {
    const int kbase = kb * 32;
    const int cur   = kb & 1;
    if (kb + 1 < SS / 32) issueKV(kbase + 32, cur ^ 1);

    // ---- scores S = Q K^T (16 x 32 = two 16x16 C tiles), K-frags from LDS ----
    FragC sc[2];
#pragma unroll
    for (int ct = 0; ct < 2; ++ct) {
#pragma unroll
      for (int i = 0; i < 8; ++i) sc[ct].f[i] = 0.0f;
      int keyrow = 16 * ct + l16;   // B-fragment column = key index
#pragma unroll
      for (int kc = 0; kc < 2; ++kc) {   // head_dim = 64 -> two K32 steps
        FragAB kf;
        const unsigned char* kp = &Ksm[cur][keyrow * KROW + 64 * kc + 32 * khalf];
#pragma unroll
        for (int i = 0; i < 8; ++i)
          kf.u[i] = *(const unsigned int*)(kp + 4 * i);
        sc[ct].v = __builtin_amdgcn_wmma_f32_16x16x32_bf16(
            false, qa[kc].v, false, kf.v, (short)0, sc[ct].v, false, false);
      }
    }

    // ---- online softmax (each 16-lane half holds one full row per VGPR idx) ----
#pragma unroll
    for (int i = 0; i < 8; ++i) {
      float s0 = sc[0].f[i] * SCALE;
      float s1 = sc[1].f[i] * SCALE;
      float v = fmaxf(s0, s1);
#pragma unroll
      for (int off = 1; off < 16; off <<= 1) v = fmaxf(v, __shfl_xor(v, off, 32));
      float mnew  = fmaxf(mrow[i], v);
      float alpha = __expf(mrow[i] - mnew);
      float p0 = __expf(s0 - mnew);
      float p1 = __expf(s1 - mnew);
      float rs = p0 + p1;
#pragma unroll
      for (int off = 1; off < 16; off <<= 1) rs += __shfl_xor(rs, off, 32);
      mrow[i] = mnew;
      lrow[i] = lrow[i] * alpha + rs;
#pragma unroll
      for (int t = 0; t < 4; ++t) acc[t].f[i] *= alpha;
      int row = i + 8 * khalf;
      Plds[wave][row * 34 + l16]      = f32_to_bf16_bits(p0);
      Plds[wave][row * 34 + 16 + l16] = f32_to_bf16_bits(p1);
    }

    // ---- read P back as a 16x32 A-fragment (wave-private LDS, DS in-order) ----
    FragAB pa;
#pragma unroll
    for (int i = 0; i < 8; ++i) {
      int kk = 16 * khalf + 2 * i;
      pa.u[i] = *(const unsigned int*)&Plds[wave][l16 * 34 + kk];
    }

    // ---- O += P @ V, V-frags from LDS ----
#pragma unroll
    for (int t = 0; t < 4; ++t) {
      FragAB vf;
      const unsigned char* vp = &Vsm[cur][(16 * t + l16) * VROW + 32 * khalf];
#pragma unroll
      for (int i = 0; i < 8; ++i)
        vf.u[i] = *(const unsigned int*)(vp + 4 * i);
      acc[t].v = __builtin_amdgcn_wmma_f32_16x16x32_bf16(
          false, pa.v, false, vf.v, (short)0, acc[t].v, false, false);
    }

    wait_async0();
    __syncthreads();
  }

  // ---- normalize and store bf16 attention output [b][s][h*HD+hd] ----
#pragma unroll
  for (int t = 0; t < 4; ++t) {
    int nn = h * HDm + 16 * t + l16;
#pragma unroll
    for (int i = 0; i < 8; ++i) {
      int   s   = q0 + i + 8 * khalf;
      float val = acc[t].f[i] / lrow[i];
      O[((size_t)(b * SS + s)) * DD + nn] = f32_to_bf16_bits(val);
    }
  }
}

// ---------------------------------------------------------------- launch
extern "C" void kernel_launch(void* const* d_in, const int* in_sizes, int n_in,
                              void* d_out, int out_size, void* d_ws, size_t ws_size,
                              hipStream_t stream) {
  const float* q  = (const float*)d_in[0];
  const float* Wq = (const float*)d_in[1];
  const float* bq = (const float*)d_in[2];
  const float* Wk = (const float*)d_in[3];
  const float* bk = (const float*)d_in[4];
  const float* Wv = (const float*)d_in[5];
  const float* bv = (const float*)d_in[6];
  const float* Wo = (const float*)d_in[7];
  const float* bo = (const float*)d_in[8];
  float* out = (float*)d_out;

  char* ws = (char*)d_ws;
  unsigned short* Xbf = (unsigned short*)ws; ws += (size_t)MM * DD * 2;
  unsigned short* Wqt = (unsigned short*)ws; ws += (size_t)DD * DD * 2;
  unsigned short* Wkt = (unsigned short*)ws; ws += (size_t)DD * DD * 2;
  unsigned short* Wvt = (unsigned short*)ws; ws += (size_t)DD * DD * 2;
  unsigned short* Wot = (unsigned short*)ws; ws += (size_t)DD * DD * 2;
  unsigned short* Qbf = (unsigned short*)ws; ws += (size_t)MM * DD * 2;
  unsigned short* Kbf = (unsigned short*)ws; ws += (size_t)MM * DD * 2;
  unsigned short* Vtb = (unsigned short*)ws; ws += (size_t)MM * DD * 2;
  unsigned short* Att = (unsigned short*)ws; ws += (size_t)MM * DD * 2;

  const int nX = MM * DD;        // 4,194,304
  const int nW = DD * DD;        // 1,048,576
  cvt_f32_to_bf16<<<(nX + 255) / 256, 256, 0, stream>>>(q, Xbf, nX);
  cvt_transpose_w<<<(nW + 255) / 256, 256, 0, stream>>>(Wq, Wqt);
  cvt_transpose_w<<<(nW + 255) / 256, 256, 0, stream>>>(Wk, Wkt);
  cvt_transpose_w<<<(nW + 255) / 256, 256, 0, stream>>>(Wv, Wvt);
  cvt_transpose_w<<<(nW + 255) / 256, 256, 0, stream>>>(Wo, Wot);

  dim3 gg(MM / 64, DD / 64);     // 64 x 16 blocks, 128 threads (4 waves)
  gemm_wmma<<<gg, 128, 0, stream>>>(Xbf, Wqt, bq, Qbf, nullptr, 0);
  gemm_wmma<<<gg, 128, 0, stream>>>(Xbf, Wkt, bk, Kbf, nullptr, 0);
  gemm_wmma<<<gg, 128, 0, stream>>>(Xbf, Wvt, bv, Vtb, nullptr, 1);

  attn_wmma<<<dim3(SS / 64, BB * HH), 128, 0, stream>>>(Qbf, Kbf, Vtb, Att);

  gemm_wmma<<<gg, 128, 0, stream>>>(Att, Wot, bo, nullptr, out, 2);
}
